// CorrBlock_61357902790685
// MI455X (gfx1250) — matrix operand order repaired and checked by reading, not compiled, not run on hardware
//
#include <hip/hip_runtime.h>

#define NPTS 8192
#define CDIM 128
#define KSEL 128
#define NTILES (NPTS / 16)        // 512
#define CHUNK 2048
#define NCHUNK (NPTS / CHUNK)     // 4
#define MT_PER_CHUNK (CHUNK / 16) // 128
#define VPT (CHUNK / 256)         // values per thread in selection = 8

typedef __attribute__((ext_vector_type(16))) __bf16 v16bf;
typedef __attribute__((ext_vector_type(8)))  float  v8f;

struct Pair { float v; int i; };

__device__ __forceinline__ unsigned ordf(float f) {
  unsigned u = __float_as_uint(f);
  return (u & 0x80000000u) ? ~u : (u | 0x80000000u);
}
__device__ __forceinline__ float deordf(unsigned k) {
  unsigned u = (k & 0x80000000u) ? (k ^ 0x80000000u) : ~k;
  return __uint_as_float(u);
}
__device__ __forceinline__ unsigned short bf16_rne(float x) {
  unsigned u = __float_as_uint(x);
  unsigned r = u + 0x7fffu + ((u >> 16) & 1u);
  return (unsigned short)(r >> 16);
}
__device__ __forceinline__ float bf16_to_f(unsigned short h) {
  return __uint_as_float(((unsigned)h) << 16);
}
__device__ __forceinline__ bool pair_gt(float av, int ai, float bv, int bi) {
  return (av > bv) || (av == bv && ai < bi);
}

// ---------------------------------------------------------------------------
// K0: zero the group-norm statistic accumulators (48 floats used)
// ---------------------------------------------------------------------------
__global__ void cb_zero_stats(float* __restrict__ stats) {
  if (threadIdx.x < 64) stats[threadIdx.x] = 0.f;
}

// ---------------------------------------------------------------------------
// Kpack: split fmap1/fmap2 into bf16 hi/lo fragments laid out exactly as the
// CDNA5 WMMA 16-bit A-matrix (fmap1 rows) / B-matrix (fmap2 cols) expect.
// One thread per (tile, kchunk, lane); each writes 16 elems x {A,B}x{hi,lo}.
// ---------------------------------------------------------------------------
__global__ void __launch_bounds__(256) cb_pack_kernel(
                               const float* __restrict__ fmap1,
                               const float* __restrict__ fmap2,
                               unsigned short* __restrict__ Ahi,
                               unsigned short* __restrict__ Alo,
                               unsigned short* __restrict__ Bhi,
                               unsigned short* __restrict__ Blo) {
  int g = blockIdx.x * blockDim.x + threadIdx.x; // 0..65535
  int lane = g & 31;
  int kc = (g >> 5) & 3;
  int tile = g >> 7;
  int base = g * 16;
#pragma unroll
  for (int e = 0; e < 16; ++e) {
    // A fragment (ISA 16-bit A 16x32 layout)
    int cA, nA;
    if (lane < 16) { nA = tile * 16 + lane;      cA = kc * 32 + (e < 8 ? e      : 16 + (e - 8)); }
    else           { nA = tile * 16 + lane - 16; cA = kc * 32 + (e < 8 ? 8 + e  : 24 + (e - 8)); }
    float x = fmap1[cA * NPTS + nA];
    unsigned short h = bf16_rne(x);
    Ahi[base + e] = h;
    Alo[base + e] = bf16_rne(x - bf16_to_f(h));
    // B fragment (32x16, same K order as A: lanes 0-15 -> K 0..7,16..23, lanes 16-31 -> K 8..15,24..31)
    int cB, mB;
    if (lane < 16) { mB = tile * 16 + lane;      cB = kc * 32 + (e < 8 ? e      : 16 + (e - 8)); }
    else           { mB = tile * 16 + lane - 16; cB = kc * 32 + (e < 8 ? 8 + e  : 24 + (e - 8)); }
    float y = fmap2[cB * NPTS + mB];
    unsigned short hb = bf16_rne(y);
    Bhi[base + e] = hb;
    Blo[base + e] = bf16_rne(y - bf16_to_f(hb));
  }
}

// ---------------------------------------------------------------------------
// K1: fused correlation GEMM (bf16-split WMMA, ~fp32 accurate) + exact
// per-row top-128 (radix select per 2048-chunk + bitonic sort/merge).
// Block = 256 threads (8 wave32). One block owns 16 rows of corr.
// LDS-limited to 2 blocks/WGP anyway, so give the allocator full VGPRs
// (launch_bounds min-blocks 1) to kill the scratch spills seen in round 1.
// ---------------------------------------------------------------------------
__global__ void __launch_bounds__(256, 1) cb_corr_topk_kernel(
                                    const unsigned short* __restrict__ Ahi_,
                                    const unsigned short* __restrict__ Alo_,
                                    const unsigned short* __restrict__ Bhi_,
                                    const unsigned short* __restrict__ Blo_,
                                    float* __restrict__ tcorr,
                                    int* __restrict__ tidx) {
  extern __shared__ unsigned char smem[];
  float*    corrbuf = (float*)smem;                    // 16*CHUNK f32
  Pair*     cand    = (Pair*)(corrbuf + 16 * CHUNK);   // 16*128
  Pair*     sbuf    = cand + 16 * KSEL;                // 128
  Pair*     mbuf    = sbuf + KSEL;                     // 256
  unsigned* hist    = (unsigned*)(mbuf + 2 * KSEL);    // 256
  int*      tsum    = (int*)(hist + 256);              // 256
  int*      sint    = tsum + 256;                      // [0]=byte [1]=need [2]=cnt

  const v16bf* Ah = (const v16bf*)Ahi_;
  const v16bf* Al = (const v16bf*)Alo_;
  const v16bf* Bh = (const v16bf*)Bhi_;
  const v16bf* Bl = (const v16bf*)Blo_;

  const int tid  = threadIdx.x;
  const int lane = tid & 31;
  const int wave = tid >> 5;
  const int tile = blockIdx.x;
  const float scale = 0.08838834764831845f; // 1/sqrt(128)

  v16bf a_hi[4], a_lo[4];
#pragma unroll
  for (int kc = 0; kc < 4; ++kc) {
    a_hi[kc] = Ah[(tile * 4 + kc) * 32 + lane];
    a_lo[kc] = Al[(tile * 4 + kc) * 32 + lane];
  }

  for (int ch = 0; ch < NCHUNK; ++ch) {
    // ---- GEMM phase: each wave computes 16 of the 128 m-tiles ----
    for (int q = 0; q < MT_PER_CHUNK / 8; ++q) {
      int mt  = wave + 8 * q;
      int gmt = ch * MT_PER_CHUNK + mt;
      const v16bf* bhp = &Bh[(gmt * 4) * 32 + lane];
      const v16bf* blp = &Bl[(gmt * 4) * 32 + lane];
      __builtin_prefetch(&Bh[((gmt + 8) * 4) * 32 + lane], 0, 3);
      v16bf b_hi[4], b_lo[4];
#pragma unroll
      for (int kc = 0; kc < 4; ++kc) { b_hi[kc] = bhp[kc * 32]; b_lo[kc] = blp[kc * 32]; }
      v8f acc = {};
#pragma unroll
      for (int kc = 0; kc < 4; ++kc)
        acc = __builtin_amdgcn_wmma_f32_16x16x32_bf16(false, a_hi[kc], false, b_hi[kc], (short)0, acc, false, false);
#pragma unroll
      for (int kc = 0; kc < 4; ++kc)
        acc = __builtin_amdgcn_wmma_f32_16x16x32_bf16(false, a_hi[kc], false, b_lo[kc], (short)0, acc, false, false);
#pragma unroll
      for (int kc = 0; kc < 4; ++kc)
        acc = __builtin_amdgcn_wmma_f32_16x16x32_bf16(false, a_lo[kc], false, b_hi[kc], (short)0, acc, false, false);
      int col = mt * 16 + (lane & 15);
      int rowoff = (lane < 16) ? 0 : 8; // C/D layout: VGPR i -> M=i (lanes 0-15) / M=i+8
#pragma unroll
      for (int i = 0; i < 8; ++i)
        corrbuf[(rowoff + i) * CHUNK + col] = acc[i] * scale;
    }
    __syncthreads();

    // ---- selection phase: exact top-128 per row of this chunk, merge ----
    for (int r = 0; r < 16; ++r) {
      const float* rowv = corrbuf + r * CHUNK;
      unsigned prefix = 0, pmask = 0;
      int need = KSEL;
      for (int pass = 0; pass < 4; ++pass) {
        int shift = 24 - 8 * pass;
        hist[tid] = 0;
        __syncthreads();
#pragma unroll
        for (int q = 0; q < VPT; ++q) {
          int p = tid * VPT + q;
          unsigned k = ordf(rowv[p]);
          if ((k & pmask) == prefix) atomicAdd(&hist[(k >> shift) & 255], 1u);
        }
        __syncthreads();
        if (tid == 0) {
          int cum = 0, b = 255;
          for (; b > 0; --b) { int h2 = (int)hist[b]; if (cum + h2 >= need) break; cum += h2; }
          sint[0] = b; sint[1] = need - cum;
        }
        __syncthreads();
        prefix |= ((unsigned)sint[0]) << shift;
        pmask  |= 255u << shift;
        need = sint[1];
        __syncthreads();
      }
      unsigned tau = prefix; // key of the 128th largest; take `need` of == tau
      if (tid == 0) sint[2] = 0;
      __syncthreads();
      int eqf[VPT]; int loc = 0;
#pragma unroll
      for (int q = 0; q < VPT; ++q) {
        int p = tid * VPT + q;
        unsigned k = ordf(rowv[p]);
        if (k > tau) {
          int pos = atomicAdd(&sint[2], 1);
          if (pos < KSEL) { sbuf[pos].v = rowv[p]; sbuf[pos].i = ch * CHUNK + p; }
        }
        eqf[q] = (k == tau);
        loc += eqf[q];
      }
      __syncthreads();
      int G = sint[2];
      tsum[tid] = loc;
      __syncthreads();
      if (tid == 0) { int run = 0; for (int i2 = 0; i2 < 256; ++i2) { int tt = tsum[i2]; tsum[i2] = run; run += tt; } }
      __syncthreads();
      int rbase = tsum[tid]; // stable by position -> ties pick lowest index (jax order)
#pragma unroll
      for (int q = 0; q < VPT; ++q) {
        if (eqf[q]) {
          if (rbase < need) {
            int p = tid * VPT + q;
            sbuf[G + rbase].v = rowv[p];
            sbuf[G + rbase].i = ch * CHUNK + p;
          }
          rbase++;
        }
      }
      __syncthreads();
      // bitonic sort sbuf[0..127] descending (value desc, idx asc)
      for (int k2 = 2; k2 <= KSEL; k2 <<= 1) {
        for (int j = k2 >> 1; j > 0; j >>= 1) {
          if (tid < KSEL) {
            int i2 = tid, l = i2 ^ j;
            if (l > i2) {
              Pair a = sbuf[i2], bq = sbuf[l];
              bool up = ((i2 & k2) == 0);
              bool agtb = pair_gt(a.v, a.i, bq.v, bq.i);
              if (up ? !agtb : agtb) { sbuf[i2] = bq; sbuf[l] = a; }
            }
          }
          __syncthreads();
        }
      }
      if (ch == 0) {
        if (tid < KSEL) cand[r * KSEL + tid] = sbuf[tid];
        __syncthreads();
      } else {
        if (tid < KSEL) { mbuf[tid] = cand[r * KSEL + tid]; mbuf[KSEL + tid] = sbuf[KSEL - 1 - tid]; }
        __syncthreads();
        for (int j = KSEL; j > 0; j >>= 1) { // bitonic merge of 256, descending
          if (tid < KSEL) {
            int i2 = ((tid / j) * (j << 1)) + (tid % j);
            int l = i2 + j;
            Pair a = mbuf[i2], bq = mbuf[l];
            if (pair_gt(bq.v, bq.i, a.v, a.i)) { mbuf[i2] = bq; mbuf[l] = a; }
          }
          __syncthreads();
        }
        if (tid < KSEL) cand[r * KSEL + tid] = mbuf[tid];
        __syncthreads();
      }
    }
    __syncthreads();
  }
  for (int r = 0; r < 16; ++r) {
    if (tid < KSEL) {
      Pair pq = cand[r * KSEL + tid];
      int row = tile * 16 + r;
      tcorr[row * KSEL + tid] = pq.v;
      tidx[row * KSEL + tid]  = pq.i;
    }
  }
}

// ---------------------------------------------------------------------------
// K2: per-point geometry: 3-level voxel pooling feats[0:81], coarse scatter-
// max + moves + voxel_corr (vox MLP input), KNN top-32, and global GN-stat
// accumulation for the vox/knn conv1 outputs. Block = 128 threads (K=128).
// ---------------------------------------------------------------------------
__global__ void __launch_bounds__(128) cb_point_kernel(
                                const float* __restrict__ tcorr, const int* __restrict__ tidx,
                                const float* __restrict__ xyz2, const float* __restrict__ coords,
                                const float* __restrict__ vox_w1, const float* __restrict__ vox_b1,
                                const float* __restrict__ knn_w1, const float* __restrict__ knn_b1,
                                float* __restrict__ feats, float* __restrict__ vin,
                                float* __restrict__ kin, float* __restrict__ stats) {
  __shared__ float px[128], py[128], pz[128], lc[128];
  __shared__ float cadd[27], ccnt[27];
  __shared__ unsigned long long ckey[27];
  __shared__ float mvx[27], mvy[27], mvz[27];
  __shared__ int   oob[27];
  __shared__ float vadd[28], vcnt[28];
  __shared__ float vcor[27];
  __shared__ float sdv[128];
  __shared__ int   sdi[128];
  __shared__ float kcv[32], kxv[32], kyv[32], kzv[32];
  __shared__ float gsum[8], gsq[8];

  int n = blockIdx.x, t = threadIdx.x;
  int gi = tidx[n * KSEL + t];
  float cv = tcorr[n * KSEL + t];
  px[t] = xyz2[gi * 3 + 0]; py[t] = xyz2[gi * 3 + 1]; pz[t] = xyz2[gi * 3 + 2];
  lc[t] = cv;
  float c0 = coords[n * 3 + 0], c1 = coords[n * 3 + 1], c2 = coords[n * 3 + 2];
  __syncthreads();

  // multi-level voxel average pooling
  for (int lev = 0; lev < 3; ++lev) {
    float r = 0.25f * (float)(1 << lev);
    if (t < 27) { cadd[t] = 0.f; ccnt[t] = 0.f; }
    __syncthreads();
    float d0 = rintf((px[t] - c0) / r), d1 = rintf((py[t] - c1) / r), d2 = rintf((pz[t] - c2) / r);
    bool valid = fabsf(d0) <= 1.f && fabsf(d1) <= 1.f && fabsf(d2) <= 1.f;
    if (valid) {
      int cube = (int)((d0 + 1.f) * 9.f + (d1 + 1.f) * 3.f + (d2 + 1.f));
      atomicAdd(&cadd[cube], cv);
      atomicAdd(&ccnt[cube], 1.f);
    }
    __syncthreads();
    if (t < 27) {
      float cnt = fminf(fmaxf(ccnt[t], 1.f), (float)NPTS);
      feats[(lev * 27 + t) * NPTS + n] = cadd[t] / cnt;
    }
    __syncthreads();
  }

  // coarse level: scatter-max packed as (ordered(val)<<32 | (127-k)) -> max
  // value, first occurrence (lowest k) on ties, like argmax.
  if (t < 27) ckey[t] = 0ull;
  __syncthreads();
  {
    float d0 = rintf((px[t] - c0) * 0.25f), d1 = rintf((py[t] - c1) * 0.25f), d2 = rintf((pz[t] - c2) * 0.25f);
    bool vld = fabsf(d0) <= 1.5f && fabsf(d1) <= 1.5f && fabsf(d2) <= 1.5f;
    if (vld) {
      int j = (int)((d0 + 1.f) * 9.f + (d1 + 1.f) * 3.f + (d2 + 1.f));
      unsigned long long key = (((unsigned long long)ordf(cv)) << 32) | (unsigned)(127 - t);
      atomicMax(&ckey[j], key);
    }
  }
  __syncthreads();
  if (t < 27) {
    unsigned long long key = ckey[t];
    int kidx = 0, ob;
    if (key == 0ull) ob = 1;
    else {
      float val = deordf((unsigned)(key >> 32));
      kidx = 127 - (int)(unsigned)(key & 0xffffffffull);
      ob = (val <= 0.f) ? 1 : 0;
      if (ob) kidx = 0;
    }
    oob[t] = ob;
    float cx = px[kidx], cy = py[kidx], cz = pz[kidx];
    int bx = (t / 9) % 3 - 1, by = (t / 3) % 3 - 1, bz = t % 3 - 1;
    float v0 = c0 + bx * 4.f, v1 = c1 + by * 4.f, v2 = c2 + bz * 4.f;
    mvx[t] = fminf(fmaxf(cx - v0, -1.f), 1.f) + v0;
    mvy[t] = fminf(fmaxf(cy - v1, -1.f), 1.f) + v1;
    mvz[t] = fminf(fmaxf(cz - v2, -1.f), 1.f) + v2;
  }
  if (t < 28) { vadd[t] = 0.f; vcnt[t] = 0.f; }
  __syncthreads();
  int isc = 27;
  for (int j = 0; j < 27; ++j) {
    bool rep = (!oob[j]) && fabsf(px[t] - mvx[j]) <= 1.f && fabsf(py[t] - mvy[j]) <= 1.f &&
               fabsf(pz[t] - mvz[j]) <= 1.f;
    if (rep) isc = j; // later j overwrites (torch order)
  }
  atomicAdd(&vadd[isc], cv);
  atomicAdd(&vcnt[isc], 1.f);
  __syncthreads();
  if (t < 27) {
    float cnt = fminf(fmaxf(vcnt[t], 1.f), (float)NPTS);
    float vc = vadd[t] / cnt;
    vcor[t] = vc;
    vin[0 * (NPTS * 27) + n * 27 + t] = vc;
    vin[1 * (NPTS * 27) + n * 27 + t] = mvx[t];
    vin[2 * (NPTS * 27) + n * 27 + t] = mvy[t];
    vin[3 * (NPTS * 27) + n * 27 + t] = mvz[t];
  }

  // knn: sort by (dist asc, idx asc) == descending on -dist with idx-asc ties
  float dx = px[t] - c0, dy = py[t] - c1, dz = pz[t] - c2;
  sdv[t] = -(dx * dx + dy * dy + dz * dz);
  sdi[t] = t;
  __syncthreads();
  for (int k2 = 2; k2 <= 128; k2 <<= 1) {
    for (int j = k2 >> 1; j > 0; j >>= 1) {
      int l = t ^ j;
      if (l > t) {
        float av = sdv[t], bv2 = sdv[l];
        int ai = sdi[t], bi2 = sdi[l];
        bool up = ((t & k2) == 0);
        bool agtb = pair_gt(av, ai, bv2, bi2);
        if (up ? !agtb : agtb) { sdv[t] = bv2; sdi[t] = bi2; sdv[l] = av; sdi[l] = ai; }
      }
      __syncthreads();
    }
  }
  if (t < 32) {
    int nb = sdi[t];
    float k0 = lc[nb], k1 = px[nb] - c0, k2v = py[nb] - c1, k3 = pz[nb] - c2;
    kcv[t] = k0; kxv[t] = k1; kyv[t] = k2v; kzv[t] = k3;
    kin[0 * (NPTS * 32) + n * 32 + t] = k0;
    kin[1 * (NPTS * 32) + n * 32 + t] = k1;
    kin[2 * (NPTS * 32) + n * 32 + t] = k2v;
    kin[3 * (NPTS * 32) + n * 32 + t] = k3;
  }

  // vox GN stats (8 groups of 4 channels)
  if (t < 8) { gsum[t] = 0.f; gsq[t] = 0.f; }
  __syncthreads();
  for (int id = t; id < 32 * 27; id += 128) {
    int c = id / 27, j = id - c * 27;
    float h = vox_b1[c] + vox_w1[c * 4 + 0] * vcor[j] + vox_w1[c * 4 + 1] * mvx[j] +
              vox_w1[c * 4 + 2] * mvy[j] + vox_w1[c * 4 + 3] * mvz[j];
    atomicAdd(&gsum[c >> 2], h);
    atomicAdd(&gsq[c >> 2], h * h);
  }
  __syncthreads();
  if (t < 8) { atomicAdd(&stats[0 + t], gsum[t]); atomicAdd(&stats[8 + t], gsq[t]); }
  __syncthreads();
  // knn GN stats (8 groups of 8 channels)
  if (t < 8) { gsum[t] = 0.f; gsq[t] = 0.f; }
  __syncthreads();
  for (int id = t; id < 64 * 32; id += 128) {
    int c = id >> 5, j = id & 31;
    float h = knn_b1[c] + knn_w1[c * 4 + 0] * kcv[j] + knn_w1[c * 4 + 1] * kxv[j] +
              knn_w1[c * 4 + 2] * kyv[j] + knn_w1[c * 4 + 3] * kzv[j];
    atomicAdd(&gsum[c >> 3], h);
    atomicAdd(&gsq[c >> 3], h * h);
  }
  __syncthreads();
  if (t < 8) { atomicAdd(&stats[16 + t], gsum[t]); atomicAdd(&stats[24 + t], gsq[t]); }
}

// ---------------------------------------------------------------------------
// K3: apply vox/knn GN+PReLU (recompute conv1 from stored inputs), vox conv2
// -> feats[81:108], assemble vfeat, out conv1 + out-GN stats, knn max-pool.
// ---------------------------------------------------------------------------
__global__ void __launch_bounds__(128) cb_mid_kernel(
                              const float* __restrict__ vin, const float* __restrict__ kin,
                              const float* __restrict__ feats_,
                              const float* __restrict__ vox_w1, const float* __restrict__ vox_b1,
                              const float* __restrict__ vox_g, const float* __restrict__ vox_bt,
                              const float* __restrict__ vox_pr, const float* __restrict__ vox_w2,
                              const float* __restrict__ vox_b2,
                              const float* __restrict__ knn_w1, const float* __restrict__ knn_b1,
                              const float* __restrict__ knn_g, const float* __restrict__ knn_bt,
                              const float* __restrict__ knn_pr,
                              const float* __restrict__ out_w1, const float* __restrict__ out_b1,
                              float* __restrict__ hout, float* __restrict__ kfout,
                              float* __restrict__ stats) {
  __shared__ float vcor[27], mx[27], my[27], mz[27];
  __shared__ float av[32 * 27];
  __shared__ float vf[108];
  __shared__ float kcv[32], kxv[32], kyv[32], kzv[32];
  __shared__ float ak[64 * 32];
  __shared__ float gsum[8], gsq[8];
  int n = blockIdx.x, t = threadIdx.x;
  if (t < 27) {
    vcor[t] = vin[0 * (NPTS * 27) + n * 27 + t];
    mx[t]   = vin[1 * (NPTS * 27) + n * 27 + t];
    my[t]   = vin[2 * (NPTS * 27) + n * 27 + t];
    mz[t]   = vin[3 * (NPTS * 27) + n * 27 + t];
  }
  if (t < 32) {
    kcv[t] = kin[0 * (NPTS * 32) + n * 32 + t];
    kxv[t] = kin[1 * (NPTS * 32) + n * 32 + t];
    kyv[t] = kin[2 * (NPTS * 32) + n * 32 + t];
    kzv[t] = kin[3 * (NPTS * 32) + n * 32 + t];
  }
  for (int i = t; i < 81; i += 128) vf[i] = feats_[i * NPTS + n];
  if (t < 8) { gsum[t] = 0.f; gsq[t] = 0.f; }
  __syncthreads();

  const float VOXC = 4.f * 27.f * (float)NPTS;
  float vpr = vox_pr[0];
  for (int id = t; id < 32 * 27; id += 128) {
    int c = id / 27, j = id - c * 27;
    float h = vox_b1[c] + vox_w1[c * 4 + 0] * vcor[j] + vox_w1[c * 4 + 1] * mx[j] +
              vox_w1[c * 4 + 2] * my[j] + vox_w1[c * 4 + 3] * mz[j];
    int g = c >> 2;
    float mean = stats[0 + g] / VOXC;
    float var = stats[8 + g] / VOXC - mean * mean;
    float y = (h - mean) * rsqrtf(var + 1e-5f) * vox_g[c] + vox_bt[c];
    av[id] = y >= 0.f ? y : vpr * y;
  }
  __syncthreads();
  if (t < 27) {
    float h2 = vox_b2[0];
    for (int c = 0; c < 32; ++c) h2 += vox_w2[c] * av[c * 27 + t];
    vf[81 + t] = h2;
  }
  __syncthreads();
  {
    float ho = out_b1[t];
    for (int i = 0; i < 108; ++i) ho += out_w1[t * 108 + i] * vf[i];
    hout[t * NPTS + n] = ho;
    atomicAdd(&gsum[t >> 4], ho);
    atomicAdd(&gsq[t >> 4], ho * ho);
  }
  __syncthreads();
  if (t < 8) { atomicAdd(&stats[32 + t], gsum[t]); atomicAdd(&stats[40 + t], gsq[t]); }

  const float KNNC = 8.f * 32.f * (float)NPTS;
  float kpr = knn_pr[0];
  for (int id = t; id < 64 * 32; id += 128) {
    int c = id >> 5, j = id & 31;
    float h = knn_b1[c] + knn_w1[c * 4 + 0] * kcv[j] + knn_w1[c * 4 + 1] * kxv[j] +
              knn_w1[c * 4 + 2] * kyv[j] + knn_w1[c * 4 + 3] * kzv[j];
    int g = c >> 3;
    float mean = stats[16 + g] / KNNC;
    float var = stats[24 + g] / KNNC - mean * mean;
    float y = (h - mean) * rsqrtf(var + 1e-5f) * knn_g[c] + knn_bt[c];
    ak[id] = y >= 0.f ? y : kpr * y;
  }
  __syncthreads();
  if (t < 64) {
    float m = ak[t * 32];
    for (int j = 1; j < 32; ++j) m = fmaxf(m, ak[t * 32 + j]);
    kfout[t * NPTS + n] = m;
  }
}

// ---------------------------------------------------------------------------
// K4: out GN+PReLU, out conv2 (64x128), knn out conv (64x64), final add.
// ---------------------------------------------------------------------------
__global__ void __launch_bounds__(128) cb_out_kernel(
                              const float* __restrict__ hout, const float* __restrict__ kfin,
                              const float* __restrict__ out_g, const float* __restrict__ out_bt,
                              const float* __restrict__ out_pr,
                              const float* __restrict__ out_w2, const float* __restrict__ out_b2,
                              const float* __restrict__ knn_ow, const float* __restrict__ knn_ob,
                              const float* __restrict__ stats, float* __restrict__ out) {
  __shared__ float a[128];
  __shared__ float kv[64];
  int n = blockIdx.x, t = threadIdx.x;
  const float OUTC = 16.f * (float)NPTS;
  float opr = out_pr[0];
  {
    float h = hout[t * NPTS + n];
    int g = t >> 4;
    float mean = stats[32 + g] / OUTC;
    float var = stats[40 + g] / OUTC - mean * mean;
    float y = (h - mean) * rsqrtf(var + 1e-5f) * out_g[t] + out_bt[t];
    a[t] = y >= 0.f ? y : opr * y;
  }
  if (t < 64) kv[t] = kfin[t * NPTS + n];
  __syncthreads();
  if (t < 64) {
    float vo = out_b2[t];
    for (int i = 0; i < 128; ++i) vo += out_w2[t * 128 + i] * a[i];
    float ko = knn_ob[t];
    for (int i = 0; i < 64; ++i) ko += knn_ow[t * 64 + i] * kv[i];
    out[t * NPTS + n] = vo + ko;
  }
}

// ---------------------------------------------------------------------------
// Host launch
// ---------------------------------------------------------------------------
static constexpr size_t SZ_PACK  = (size_t)NTILES * 4 * 32 * 16 * 2; // 2 MB each
static constexpr size_t OFF_TCORR = 0;
static constexpr size_t OFF_TIDX  = OFF_TCORR + (size_t)NPTS * KSEL * 4;
static constexpr size_t OFF_FEATS = OFF_TIDX + (size_t)NPTS * KSEL * 4;
static constexpr size_t OFF_VIN   = OFF_FEATS + (size_t)108 * NPTS * 4;
static constexpr size_t OFF_KIN   = OFF_VIN + (size_t)4 * NPTS * 27 * 4;
static constexpr size_t OFF_HOUT  = OFF_KIN + (size_t)4 * NPTS * 32 * 4;
static constexpr size_t OFF_KF    = OFF_HOUT + (size_t)128 * NPTS * 4;
static constexpr size_t OFF_APHI  = OFF_KF + (size_t)64 * NPTS * 4;
static constexpr size_t OFF_APLO  = OFF_APHI + SZ_PACK;
static constexpr size_t OFF_BPHI  = OFF_APLO + SZ_PACK;
static constexpr size_t OFF_BPLO  = OFF_BPHI + SZ_PACK;
static constexpr size_t OFF_STATS = OFF_BPLO + SZ_PACK;

static constexpr int K1_LDS = 16 * CHUNK * 4   // corrbuf
                            + 16 * KSEL * 8    // cand
                            + KSEL * 8         // sbuf
                            + 2 * KSEL * 8     // mbuf
                            + 256 * 4          // hist
                            + 256 * 4          // tsum
                            + 4 * 4;           // sint

extern "C" void kernel_launch(void* const* d_in, const int* in_sizes, int n_in,
                              void* d_out, int out_size, void* d_ws, size_t ws_size,
                              hipStream_t stream) {
  const float* fmap1  = (const float*)d_in[0];
  const float* fmap2  = (const float*)d_in[1];
  const float* xyz2   = (const float*)d_in[2];
  const float* coords = (const float*)d_in[3];
  const float* out_w1 = (const float*)d_in[4];
  const float* out_b1 = (const float*)d_in[5];
  const float* out_g  = (const float*)d_in[6];
  const float* out_bt = (const float*)d_in[7];
  const float* out_pr = (const float*)d_in[8];
  const float* out_w2 = (const float*)d_in[9];
  const float* out_b2 = (const float*)d_in[10];
  const float* vox_w1 = (const float*)d_in[11];
  const float* vox_b1 = (const float*)d_in[12];
  const float* vox_g  = (const float*)d_in[13];
  const float* vox_bt = (const float*)d_in[14];
  const float* vox_pr = (const float*)d_in[15];
  const float* vox_w2 = (const float*)d_in[16];
  const float* vox_b2 = (const float*)d_in[17];
  const float* knn_w1 = (const float*)d_in[18];
  const float* knn_b1 = (const float*)d_in[19];
  const float* knn_g  = (const float*)d_in[20];
  const float* knn_bt = (const float*)d_in[21];
  const float* knn_pr = (const float*)d_in[22];
  const float* knn_ow = (const float*)d_in[23];
  const float* knn_ob = (const float*)d_in[24];
  (void)in_sizes; (void)n_in; (void)out_size; (void)ws_size;

  unsigned char* ws = (unsigned char*)d_ws;
  float* tcorr = (float*)(ws + OFF_TCORR);
  int*   tidx  = (int*)(ws + OFF_TIDX);
  float* feats = (float*)(ws + OFF_FEATS);
  float* vin   = (float*)(ws + OFF_VIN);
  float* kin   = (float*)(ws + OFF_KIN);
  float* hout  = (float*)(ws + OFF_HOUT);
  float* kf    = (float*)(ws + OFF_KF);
  unsigned short* Ahi = (unsigned short*)(ws + OFF_APHI);
  unsigned short* Alo = (unsigned short*)(ws + OFF_APLO);
  unsigned short* Bhi = (unsigned short*)(ws + OFF_BPHI);
  unsigned short* Blo = (unsigned short*)(ws + OFF_BPLO);
  float* stats = (float*)(ws + OFF_STATS);

  (void)hipFuncSetAttribute(reinterpret_cast<const void*>(cb_corr_topk_kernel),
                            hipFuncAttributeMaxDynamicSharedMemorySize, K1_LDS);

  cb_zero_stats<<<1, 64, 0, stream>>>(stats);
  cb_pack_kernel<<<256, 256, 0, stream>>>(fmap1, fmap2, Ahi, Alo, Bhi, Blo);
  cb_corr_topk_kernel<<<NTILES, 256, K1_LDS, stream>>>(Ahi, Alo, Bhi, Blo, tcorr, tidx);
  cb_point_kernel<<<NPTS, 128, 0, stream>>>(tcorr, tidx, xyz2, coords, vox_w1, vox_b1,
                                            knn_w1, knn_b1, feats, vin, kin, stats);
  cb_mid_kernel<<<NPTS, 128, 0, stream>>>(vin, kin, feats, vox_w1, vox_b1, vox_g, vox_bt,
                                          vox_pr, vox_w2, vox_b2, knn_w1, knn_b1, knn_g,
                                          knn_bt, knn_pr, out_w1, out_b1, hout, kf, stats);
  cb_out_kernel<<<NPTS, 128, 0, stream>>>(hout, kf, out_g, out_bt, out_pr, out_w2, out_b2,
                                          knn_ow, knn_ob, stats, (float*)d_out);
}
